// ActionNetwork_30906584662095
// MI455X (gfx1250) — compile-verified, matrix operand order checked by reading, not run on hardware
//
#include <hip/hip_runtime.h>

// ---------------------------------------------------------------------------
// CDNA5 / gfx1250 implementation of the ActionNetwork reference.
// Conv stages: implicit-GEMM, v_wmma_f32_16x16x32_f16, LDS-staged A slabs,
// 2x cout register blocking per wave, global_prefetch of next slab.
// Head (bilinear chain / MLP / VQ): tiny scalar kernels (<<1% of FLOPs).
// ---------------------------------------------------------------------------

typedef _Float16 half_t;
typedef half_t v16h __attribute__((ext_vector_type(16)));
typedef float  v8f  __attribute__((ext_vector_type(8)));

#define LRELU_NEG 0.2f
#define BN_EPS    1e-5f

// ---------------- problem constants ----------------
constexpr int NIMG = 64;           // B*T
constexpr int C0 = 64, C1 = 128;
constexpr int H0 = 96, W0 = 96;
constexpr int H1 = 48, W1 = 48;
constexpr int AD = 64, NA = 256;

constexpr size_t EL_A = (size_t)NIMG * H0 * W0 * C0;   // 37,748,736 f16
constexpr size_t EL_B = (size_t)NIMG * H0 * W0 * C1;   // 75,497,472 f16

// ---------------- workspace layout (bytes) ----------------
constexpr size_t OFF_BUFA   = 0;
constexpr size_t OFF_BUFB   = OFF_BUFA + EL_A * 2;
constexpr size_t OFF_BUFC   = OFF_BUFB + EL_B * 2;
constexpr size_t OFF_RB1W1P = OFF_BUFC + EL_B * 2;
constexpr size_t OFF_RB1W2P = OFF_RB1W1P + 9ull * C1 * C0 * 2;
constexpr size_t OFF_RB1WSP = OFF_RB1W2P + 9ull * C1 * C1 * 2;
constexpr size_t OFF_RB2W1P = OFF_RB1WSP + (size_t)C1 * C0 * 2;
constexpr size_t OFF_RB2W2P = OFF_RB2W1P + 9ull * C0 * C1 * 2;
constexpr size_t OFF_RB2WSP = OFF_RB2W2P + 9ull * C0 * C0 * 2;
constexpr size_t OFF_HEAD   = (OFF_RB2WSP + (size_t)C0 * C1 * 2 + 255) & ~(size_t)255;
constexpr size_t OFF_FEATX  = OFF_HEAD;
constexpr size_t OFF_FEATNX = OFF_FEATX  + 64ull * 64 * 4;
constexpr size_t OFF_ABUF   = OFF_FEATNX + 64ull * 64 * 4;
constexpr size_t OFF_PB1    = OFF_ABUF   + 64ull * 64 * 4;
constexpr size_t OFF_PB2    = OFF_PB1    + 64ull * 128 * 4;
constexpr size_t OFF_Z1     = OFF_PB2    + 64ull * 128 * 4;
constexpr size_t OFF_Z2     = OFF_Z1     + 64ull * 128 * 4;
constexpr size_t OFF_ZF     = OFF_Z2     + 64ull * 64 * 4;

// K index of fragment element pair vv for this lane half (ISA 7.12.2 layout)
__device__ __forceinline__ int frag_kbase(int vv, int lh) {
  return ((vv >> 2) << 4) | (lh << 3) | ((vv & 3) << 1);
}

// ---------------------------------------------------------------------------
// Elementwise: (states * attention) -> NHWC f16
// ---------------------------------------------------------------------------
__global__ void __launch_bounds__(256)
premul_nhwc_kernel(const float* __restrict__ S, const float* __restrict__ A,
                   half_t* __restrict__ Y, int C, int H, int W, long long total)
{
  long long i = (long long)blockIdx.x * blockDim.x + threadIdx.x;
  if (i >= total) return;
  int c = (int)(i % C); long long r = i / C;
  int w = (int)(r % W); r /= W;
  int h = (int)(r % H); int n = (int)(r / H);
  float v = S[(((size_t)n * C + c) * H + h) * W + w] *
            A[((size_t)n * H + h) * W + w];
  Y[i] = (half_t)v;
}

// ---------------------------------------------------------------------------
// Weight packing: OIHW f32 -> [tap][cout][cin] f16 (3x3), or plain cast (1x1)
// ---------------------------------------------------------------------------
__global__ void __launch_bounds__(256)
pack3x3_kernel(const float* __restrict__ Ws, half_t* __restrict__ Wd, int Cout, int Cin)
{
  int i = blockIdx.x * blockDim.x + threadIdx.x;
  int total = 9 * Cout * Cin;
  if (i >= total) return;
  int ci = i % Cin; int r = i / Cin;
  int co = r % Cout; int tap = r / Cout;
  int ky = tap / 3, kx = tap % 3;
  Wd[i] = (half_t)Ws[(((size_t)co * Cin + ci) * 3 + ky) * 3 + kx];
}

__global__ void __launch_bounds__(256)
cast_f16_kernel(const float* __restrict__ s, half_t* __restrict__ d, int n)
{
  int i = blockIdx.x * blockDim.x + threadIdx.x;
  if (i < n) d[i] = (half_t)s[i];
}

// ---------------------------------------------------------------------------
// 3x3 conv, implicit GEMM, WMMA f16 -> f32 accum, fused bias+BN(+lrelu).
// Block = Cout threads (Cout/32 waves). Wave tile: 16 positions x 32 couts
// (2 accumulators); A slab (18 pos x 32 cin) staged in LDS at dword
// granularity and shared by all waves; next slab prefetched to L2.
// ---------------------------------------------------------------------------
__global__ void __launch_bounds__(128)
conv3x3_wmma(const half_t* __restrict__ X,    // NHWC, Cin
             const half_t* __restrict__ Wp,   // [tap][Cout][Cin] f16
             const float* __restrict__ cbias,
             const float* __restrict__ gamma,
             const float* __restrict__ beta,
             half_t* __restrict__ Y,          // NHWC, Cout
             int H, int W, int Cin, int Cout, int applyLrelu)
{
  __shared__ half_t sA[18 * 32];
  const int tiles_w = W >> 4;
  int tile = blockIdx.x;
  int owt = tile % tiles_w;
  int rem = tile / tiles_w;
  int oh = rem % H;
  int n  = rem / H;
  int ow0 = owt << 4;

  int tid  = threadIdx.x;
  int lane = tid & 31;
  int wave = tid >> 5;
  int m    = lane & 15;          // A row / B col / D col index
  int lh   = lane >> 4;          // K-half selector
  int coA  = (wave << 5) + m;    // first cout tile of this wave
  int coB  = coA + 16;           // second cout tile

  const half_t* bbaseA = Wp + (size_t)coA * Cin;   // per-lane B row base
  const half_t* bbaseB = Wp + (size_t)coB * Cin;
  const size_t  tapStride = (size_t)Cout * Cin;
  const half_t* xrow0 = X + ((size_t)n * H) * (size_t)W * Cin;  // image base

  v8f acc0 = {0.f, 0.f, 0.f, 0.f, 0.f, 0.f, 0.f, 0.f};
  v8f acc1 = {0.f, 0.f, 0.f, 0.f, 0.f, 0.f, 0.f, 0.f};

  for (int cin0 = 0; cin0 < Cin; cin0 += 32) {
    for (int ky = 0; ky < 3; ++ky) {
      int ih = oh + ky - 1;
      __syncthreads();
      // cooperative dword staging: 18 positions x 16 dwords (32 cin)
      for (int i = tid; i < 18 * 16; i += blockDim.x) {
        int pos = i >> 4;
        int cp  = i & 15;
        int iw  = ow0 - 1 + pos;
        unsigned v = 0u;
        if ((unsigned)ih < (unsigned)H && (unsigned)iw < (unsigned)W)
          v = *(const unsigned*)(xrow0 + ((size_t)ih * W + iw) * Cin + cin0 + 2 * cp);
        ((unsigned*)sA)[i] = v;
      }
      // prefetch next slab (ky+1 row, or next cin block) into the cache
      {
        int kyn = (ky == 2) ? 0 : ky + 1;
        int cin0n = (ky == 2) ? cin0 + 32 : cin0;
        int ihn = oh + kyn - 1;
        if (cin0n < Cin && tid < 18) {
          int iwn = ow0 - 1 + tid;
          if ((unsigned)ihn < (unsigned)H && (unsigned)iwn < (unsigned)W)
            __builtin_prefetch(xrow0 + ((size_t)ihn * W + iwn) * Cin + cin0n, 0, 3);
        }
      }
      __syncthreads();
      for (int kx = 0; kx < 3; ++kx) {
        v16h a, b0, b1;
        const half_t* arow  = &sA[(m + kx) << 5];
        const size_t  toff  = (size_t)(ky * 3 + kx) * tapStride + cin0;
        const half_t* browA = bbaseA + toff;
        const half_t* browB = bbaseB + toff;
        #pragma unroll
        for (int vv = 0; vv < 8; ++vv) {
          int kb = frag_kbase(vv, lh);
          a[2 * vv]      = arow[kb];
          a[2 * vv + 1]  = arow[kb + 1];
          b0[2 * vv]     = browA[kb];
          b0[2 * vv + 1] = browA[kb + 1];
          b1[2 * vv]     = browB[kb];
          b1[2 * vv + 1] = browB[kb + 1];
        }
        acc0 = __builtin_amdgcn_wmma_f32_16x16x32_f16(false, a, false, b0,
                                                      (short)0, acc0, false, false);
        acc1 = __builtin_amdgcn_wmma_f32_16x16x32_f16(false, a, false, b1,
                                                      (short)0, acc1, false, false);
      }
    }
  }

  const float inv = rsqrtf(1.0f + BN_EPS);
  float sA0 = gamma[coA] * inv, bA = cbias[coA] * sA0 + beta[coA];
  float sB0 = gamma[coB] * inv, bB = cbias[coB] * sB0 + beta[coB];
  half_t* ybase = Y + (((size_t)n * H + oh) * W + ow0) * Cout;
  #pragma unroll
  for (int r = 0; r < 8; ++r) {
    int mm = r + (lh << 3);
    float v0 = acc0[r] * sA0 + bA;
    float v1 = acc1[r] * sB0 + bB;
    if (applyLrelu) {
      v0 = v0 >= 0.f ? v0 : LRELU_NEG * v0;
      v1 = v1 >= 0.f ? v1 : LRELU_NEG * v1;
    }
    ybase[(size_t)mm * Cout + coA] = (half_t)v0;
    ybase[(size_t)mm * Cout + coB] = (half_t)v1;
  }
}

// ---------------------------------------------------------------------------
// 1x1 shortcut conv (WMMA) fused with residual add + leaky-ReLU.
// Same blocking: Cout threads, 32 couts per wave.
// ---------------------------------------------------------------------------
__global__ void __launch_bounds__(128)
conv1x1_add_lrelu_wmma(const half_t* __restrict__ X, const half_t* __restrict__ Wp,
                       const float* __restrict__ cbias, const half_t* __restrict__ Add,
                       half_t* __restrict__ Y, int H, int W, int Cin, int Cout)
{
  __shared__ half_t sA[16 * 32];
  const int tiles_w = W >> 4;
  int tile = blockIdx.x;
  int owt = tile % tiles_w;
  int rem = tile / tiles_w;
  int oh = rem % H;
  int n  = rem / H;
  int ow0 = owt << 4;

  int tid  = threadIdx.x;
  int lane = tid & 31;
  int wave = tid >> 5;
  int m = lane & 15, lh = lane >> 4;
  int coA = (wave << 5) + m;
  int coB = coA + 16;

  const half_t* bbaseA = Wp + (size_t)coA * Cin;
  const half_t* bbaseB = Wp + (size_t)coB * Cin;
  const half_t* xline  = X + (((size_t)n * H + oh) * W + ow0) * Cin;

  v8f acc0 = {0.f, 0.f, 0.f, 0.f, 0.f, 0.f, 0.f, 0.f};
  v8f acc1 = {0.f, 0.f, 0.f, 0.f, 0.f, 0.f, 0.f, 0.f};

  for (int cin0 = 0; cin0 < Cin; cin0 += 32) {
    __syncthreads();
    for (int i = tid; i < 16 * 16; i += blockDim.x) {
      int pos = i >> 4, cp = i & 15;
      ((unsigned*)sA)[i] =
          *(const unsigned*)(xline + (size_t)pos * Cin + cin0 + 2 * cp);
    }
    if (cin0 + 32 < Cin && tid < 16)
      __builtin_prefetch(xline + (size_t)tid * Cin + cin0 + 32, 0, 3);
    __syncthreads();
    v16h a, b0, b1;
    const half_t* arow = &sA[m << 5];
    #pragma unroll
    for (int vv = 0; vv < 8; ++vv) {
      int kb = frag_kbase(vv, lh);
      a[2 * vv]      = arow[kb];
      a[2 * vv + 1]  = arow[kb + 1];
      b0[2 * vv]     = bbaseA[cin0 + kb];
      b0[2 * vv + 1] = bbaseA[cin0 + kb + 1];
      b1[2 * vv]     = bbaseB[cin0 + kb];
      b1[2 * vv + 1] = bbaseB[cin0 + kb + 1];
    }
    acc0 = __builtin_amdgcn_wmma_f32_16x16x32_f16(false, a, false, b0,
                                                  (short)0, acc0, false, false);
    acc1 = __builtin_amdgcn_wmma_f32_16x16x32_f16(false, a, false, b1,
                                                  (short)0, acc1, false, false);
  }

  float cbA = cbias[coA], cbB = cbias[coB];
  const half_t* abase = Add + (((size_t)n * H + oh) * W + ow0) * Cout;
  half_t*       ybase = Y   + (((size_t)n * H + oh) * W + ow0) * Cout;
  #pragma unroll
  for (int r = 0; r < 8; ++r) {
    int mm = r + (lh << 3);
    size_t iA = (size_t)mm * Cout + coA;
    size_t iB = (size_t)mm * Cout + coB;
    float v0 = acc0[r] + cbA + (float)abase[iA];
    float v1 = acc1[r] + cbB + (float)abase[iB];
    v0 = v0 >= 0.f ? v0 : LRELU_NEG * v0;
    v1 = v1 >= 0.f ? v1 : LRELU_NEG * v1;
    ybase[iA] = (half_t)v0;
    ybase[iB] = (half_t)v1;
  }
}

// ---------------------------------------------------------------------------
// 2x2 average pool (NHWC f16)
// ---------------------------------------------------------------------------
__global__ void __launch_bounds__(256)
avgpool2_kernel(const half_t* __restrict__ X, half_t* __restrict__ Y,
                int H, int W, int C, long long total)
{
  long long i = (long long)blockIdx.x * blockDim.x + threadIdx.x;
  if (i >= total) return;
  int c = (int)(i % C); long long r = i / C;
  int Wo = W >> 1, Ho = H >> 1;
  int ow = (int)(r % Wo); r /= Wo;
  int oh = (int)(r % Ho); int n = (int)(r / Ho);
  size_t base = (((size_t)n * H + oh * 2) * W + ow * 2) * C + c;
  size_t rowS = (size_t)W * C;
  float s = (float)X[base] + (float)X[base + C] +
            (float)X[base + rowS] + (float)X[base + rowS + C];
  Y[i] = (half_t)(0.25f * s);
}

// ---------------------------------------------------------------------------
// Global average pool (NHWC f16 -> [n][c] f32)
// ---------------------------------------------------------------------------
__global__ void __launch_bounds__(256)
gap_kernel(const half_t* __restrict__ X, float* __restrict__ F, int HW, int C)
{
  int t = blockIdx.x * blockDim.x + threadIdx.x;
  if (t >= NIMG * C) return;
  int n = t / C, c = t % C;
  const half_t* p = X + (size_t)n * HW * C + c;
  float s = 0.f;
  for (int i = 0; i < HW; ++i) s += (float)p[(size_t)i * C];
  F[t] = s / (float)HW;
}

// ---------------------------------------------------------------------------
// Dense head: out[b,o] = (in[b,:]-sub[b,:]) @ w[o,:] + bias[o]  (+BN +lrelu)
// ---------------------------------------------------------------------------
__global__ void __launch_bounds__(256)
linear_kernel(const float* __restrict__ in, const float* __restrict__ sub,
              const float* __restrict__ w, const float* __restrict__ bias,
              const float* __restrict__ g, const float* __restrict__ be,
              float* __restrict__ out, int In, int Out, int doBN, int doLrelu)
{
  int t = blockIdx.x * blockDim.x + threadIdx.x;
  if (t >= 64 * Out) return;
  int b = t / Out, o = t % Out;
  float s = bias[o];
  for (int c = 0; c < In; ++c) {
    float x = in[b * In + c];
    if (sub) x -= sub[b * In + c];
    s += x * w[(size_t)o * In + c];
  }
  if (doBN) { s = s * (g[o] * rsqrtf(1.f + BN_EPS)) + be[o]; }
  if (doLrelu) s = s >= 0.f ? s : LRELU_NEG * s;
  out[t] = s;
}

// ---------------------------------------------------------------------------
// Bilinear: out[b,o] = sum_i a[b,i] * sum_j pw[o,i,j] * p[b,j] + pb[o]
// ---------------------------------------------------------------------------
__global__ void __launch_bounds__(256)
bilinear_kernel(const float* __restrict__ a, const float* __restrict__ p,
                const float* __restrict__ pw, const float* __restrict__ pb,
                const float* __restrict__ bng, const float* __restrict__ bnb,
                float* __restrict__ out, int I, int J, int O, int doBN)
{
  int t = blockIdx.x * blockDim.x + threadIdx.x;
  if (t >= 64 * O) return;
  int b = t / O, o = t % O;
  float acc = pb[o];
  for (int i = 0; i < I; ++i) {
    float s = 0.f;
    const float* row = pw + ((size_t)o * I + i) * J;
    for (int j = 0; j < J; ++j) s += row[j] * p[b * J + j];
    acc += a[b * I + i] * s;
  }
  if (doBN) {
    acc = acc * (bng[o] * rsqrtf(1.f + BN_EPS)) + bnb[o];
    acc = acc >= 0.f ? acc : LRELU_NEG * acc;
  }
  out[t] = acc;
}

// ---------------------------------------------------------------------------
// Vector quantizer: argmin distance, zq_st (== zq numerically), loss, idx.
// ---------------------------------------------------------------------------
__global__ void __launch_bounds__(64)
vq_kernel(const float* __restrict__ z, const float* __restrict__ emb,
          float* __restrict__ out)
{
  __shared__ float serr[64];
  int b = threadIdx.x;
  float zl[AD];
  #pragma unroll
  for (int j = 0; j < AD; ++j) zl[j] = z[b * AD + j];
  int best = 0; float bestd = 3.4e38f;
  for (int e = 0; e < NA; ++e) {
    float d = 0.f;
    for (int j = 0; j < AD; ++j) { float t = emb[e * AD + j]; d += t * t - 2.f * t * zl[j]; }
    if (d < bestd) { bestd = d; best = e; }
  }
  float err = 0.f;
  for (int j = 0; j < AD; ++j) {
    float q = emb[best * AD + j];
    out[1 + b * AD + j] = q;          // zq_st = z + sg(zq - z) == zq numerically
    float dd = q - zl[j];
    err += dd * dd;
  }
  out[1 + 64 * AD + b] = (float)best;
  serr[b] = err;
  __syncthreads();
  if (b == 0) {
    float s = 0.f;
    for (int i = 0; i < 64; ++i) s += serr[i];
    out[0] = (1.0f + 0.25f) * s / (64.0f * (float)AD);  // (BETA+1)*mean
  }
}

// ---------------------------------------------------------------------------
// Host orchestration
// ---------------------------------------------------------------------------
extern "C" void kernel_launch(void* const* d_in, const int* in_sizes, int n_in,
                              void* d_out, int out_size, void* d_ws, size_t ws_size,
                              hipStream_t stream)
{
  (void)in_sizes; (void)n_in; (void)out_size; (void)ws_size;
  char* ws = (char*)d_ws;
  half_t* bufA = (half_t*)(ws + OFF_BUFA);
  half_t* bufB = (half_t*)(ws + OFF_BUFB);
  half_t* bufC = (half_t*)(ws + OFF_BUFC);
  half_t* rb1w1p = (half_t*)(ws + OFF_RB1W1P);
  half_t* rb1w2p = (half_t*)(ws + OFF_RB1W2P);
  half_t* rb1wsp = (half_t*)(ws + OFF_RB1WSP);
  half_t* rb2w1p = (half_t*)(ws + OFF_RB2W1P);
  half_t* rb2w2p = (half_t*)(ws + OFF_RB2W2P);
  half_t* rb2wsp = (half_t*)(ws + OFF_RB2WSP);
  float* featX  = (float*)(ws + OFF_FEATX);
  float* featNX = (float*)(ws + OFF_FEATNX);
  float* abuf   = (float*)(ws + OFF_ABUF);
  float* pb1    = (float*)(ws + OFF_PB1);
  float* pb2    = (float*)(ws + OFF_PB2);
  float* z1     = (float*)(ws + OFF_Z1);
  float* z2     = (float*)(ws + OFF_Z2);
  float* zf     = (float*)(ws + OFF_ZF);

  // input pointers (setup_inputs order)
  const float* rb1_w1 = (const float*)d_in[5];
  const float* rb1_b1 = (const float*)d_in[6];
  const float* rb1_g1 = (const float*)d_in[7];
  const float* rb1_be1= (const float*)d_in[8];
  const float* rb1_w2 = (const float*)d_in[9];
  const float* rb1_b2 = (const float*)d_in[10];
  const float* rb1_g2 = (const float*)d_in[11];
  const float* rb1_be2= (const float*)d_in[12];
  const float* rb1_ws_ = (const float*)d_in[13];
  const float* rb1_bs = (const float*)d_in[14];
  const float* rb2_w1 = (const float*)d_in[15];
  const float* rb2_b1 = (const float*)d_in[16];
  const float* rb2_g1 = (const float*)d_in[17];
  const float* rb2_be1= (const float*)d_in[18];
  const float* rb2_w2 = (const float*)d_in[19];
  const float* rb2_b2 = (const float*)d_in[20];
  const float* rb2_g2 = (const float*)d_in[21];
  const float* rb2_be2= (const float*)d_in[22];
  const float* rb2_ws_ = (const float*)d_in[23];
  const float* rb2_bs = (const float*)d_in[24];
  const float* aff_w  = (const float*)d_in[25];
  const float* aff_b  = (const float*)d_in[26];
  const float* pws[4] = {(const float*)d_in[27], (const float*)d_in[29],
                         (const float*)d_in[31], (const float*)d_in[33]};
  const float* pbs[4] = {(const float*)d_in[28], (const float*)d_in[30],
                         (const float*)d_in[32], (const float*)d_in[34]};
  const float* bn_g   = (const float*)d_in[35];
  const float* bn_b   = (const float*)d_in[36];
  const float* mlp_w1 = (const float*)d_in[37];
  const float* mlp_b1 = (const float*)d_in[38];
  const float* mlp_g1 = (const float*)d_in[39];
  const float* mlp_be1= (const float*)d_in[40];
  const float* mlp_w2 = (const float*)d_in[41];
  const float* mlp_b2 = (const float*)d_in[42];
  const float* mlp_g2 = (const float*)d_in[43];
  const float* mlp_be2= (const float*)d_in[44];
  const float* mlp_w3 = (const float*)d_in[45];
  const float* mlp_b3 = (const float*)d_in[46];
  const float* emb    = (const float*)d_in[47];

  // ---- pack weights to f16 ----
  pack3x3_kernel<<<(9*C1*C0 + 255)/256, 256, 0, stream>>>(rb1_w1, rb1w1p, C1, C0);
  pack3x3_kernel<<<(9*C1*C1 + 255)/256, 256, 0, stream>>>(rb1_w2, rb1w2p, C1, C1);
  cast_f16_kernel<<<(C1*C0 + 255)/256, 256, 0, stream>>>(rb1_ws_, rb1wsp, C1*C0);
  pack3x3_kernel<<<(9*C0*C1 + 255)/256, 256, 0, stream>>>(rb2_w1, rb2w1p, C0, C1);
  pack3x3_kernel<<<(9*C0*C0 + 255)/256, 256, 0, stream>>>(rb2_w2, rb2w2p, C0, C0);
  cast_f16_kernel<<<(C0*C1 + 255)/256, 256, 0, stream>>>(rb2_ws_, rb2wsp, C0*C1);

  // ---- encoders (states, next_states) ----
  for (int e = 0; e < 2; ++e) {
    const float* S  = (const float*)d_in[e == 0 ? 0 : 2];
    const float* At = (const float*)d_in[e == 0 ? 1 : 3];
    float* feat = (e == 0) ? featX : featNX;

    long long tA = (long long)EL_A;
    premul_nhwc_kernel<<<(unsigned)((tA + 255) / 256), 256, 0, stream>>>(
        S, At, bufA, C0, H0, W0, tA);

    const unsigned g1 = NIMG * H0 * (W0 / 16);   // position tiles @96x96
    conv3x3_wmma<<<g1, C1, 0, stream>>>(bufA, rb1w1p, rb1_b1, rb1_g1, rb1_be1,
                                        bufB, H0, W0, C0, C1, 1);
    conv3x3_wmma<<<g1, C1, 0, stream>>>(bufB, rb1w2p, rb1_b2, rb1_g2, rb1_be2,
                                        bufC, H0, W0, C1, C1, 0);
    conv1x1_add_lrelu_wmma<<<g1, C1, 0, stream>>>(bufA, rb1wsp, rb1_bs, bufC,
                                                  bufB, H0, W0, C0, C1);
    long long tP = (long long)NIMG * H1 * W1 * C1;
    avgpool2_kernel<<<(unsigned)((tP + 255) / 256), 256, 0, stream>>>(
        bufB, bufC, H0, W0, C1, tP);

    const unsigned g2 = NIMG * H1 * (W1 / 16);   // position tiles @48x48
    conv3x3_wmma<<<g2, C0, 0, stream>>>(bufC, rb2w1p, rb2_b1, rb2_g1, rb2_be1,
                                        bufA, H1, W1, C1, C0, 1);
    conv3x3_wmma<<<g2, C0, 0, stream>>>(bufA, rb2w2p, rb2_b2, rb2_g2, rb2_be2,
                                        bufB, H1, W1, C0, C0, 0);
    conv1x1_add_lrelu_wmma<<<g2, C0, 0, stream>>>(bufC, rb2wsp, rb2_bs, bufB,
                                                  bufA, H1, W1, C1, C0);
    gap_kernel<<<(NIMG * C0 + 255) / 256, 256, 0, stream>>>(bufA, feat, H1 * W1, C0);
  }

  // ---- direction head: affine + bilinear chain ----
  const int Js[4] = {64, 128, 128, 128};
  const int Os[4] = {128, 128, 128, 64};
  const float* pin = featNX;
  float* pout = pb1;
  for (int k = 0; k < 4; ++k) {
    linear_kernel<<<16, 256, 0, stream>>>(featX, nullptr, aff_w + (size_t)k * 64 * 64,
                                          aff_b + (size_t)k * 64, nullptr, nullptr,
                                          abuf, 64, 64, 0, 0);
    pout = (k & 1) ? pb2 : pb1;
    bilinear_kernel<<<(64 * Os[k] + 255) / 256, 256, 0, stream>>>(
        abuf, pin, pws[k], pbs[k],
        (k < 3) ? bn_g + (size_t)k * 128 : nullptr,
        (k < 3) ? bn_b + (size_t)k * 128 : nullptr,
        pout, 64, Js[k], Os[k], (k < 3) ? 1 : 0);
    pin = pout;
  }

  // ---- final MLP (directions = p - featX fused into layer 1) ----
  linear_kernel<<<32, 256, 0, stream>>>(pin, featX, mlp_w1, mlp_b1, mlp_g1, mlp_be1,
                                        z1, 64, 128, 1, 1);
  linear_kernel<<<16, 256, 0, stream>>>(z1, nullptr, mlp_w2, mlp_b2, mlp_g2, mlp_be2,
                                        z2, 128, 64, 1, 1);
  linear_kernel<<<16, 256, 0, stream>>>(z2, nullptr, mlp_w3, mlp_b3, nullptr, nullptr,
                                        zf, 64, AD, 0, 0);

  // ---- vector quantizer ----
  vq_kernel<<<1, 64, 0, stream>>>(zf, emb, (float*)d_out);
}
